// Qwen2_5_74053826117891
// MI455X (gfx1250) — matrix operand order repaired and checked by reading, not compiled, not run on hardware
//
#include <hip/hip_runtime.h>
#include <hip/hip_bf16.h>
#include <math.h>

// ---------------- model dims ----------------
#define LAYERS  4
#define DMODEL  1024
#define NHEAD   16
#define NKV     4
#define HEADDIM 64
#define FFDIM   2816
#define VOCAB   32000
#define BATCH   4
#define SEQ     512
#define MTOK    (BATCH * SEQ)          // 2048 rows
#define KVDIM   (2 * NKV * HEADDIM)    // 512

typedef __bf16 bf16_t;
typedef __attribute__((ext_vector_type(16))) __bf16 bf16x16;
typedef __attribute__((ext_vector_type(8)))  __bf16 bf16x8;
typedef __attribute__((ext_vector_type(8)))  float  floatx8;
typedef __attribute__((ext_vector_type(4)))  unsigned int uintx4;
typedef __attribute__((ext_vector_type(4)))  int    intx4;
typedef __attribute__((ext_vector_type(8)))  int    intx8;

#define CAT16(lo, hi) __builtin_shufflevector((lo), (hi), 0, 1, 2, 3, 4, 5, 6, 7, \
                                              8, 9, 10, 11, 12, 13, 14, 15)

// =====================================================================
// Embedding gather: x[m, d] = embed[ids[m], d]   (fp32)
// =====================================================================
__global__ void embed_gather_kernel(const int* __restrict__ ids,
                                    const float* __restrict__ embed,
                                    float* __restrict__ x) {
    int idx = blockIdx.x * blockDim.x + threadIdx.x;
    if (idx >= MTOK * DMODEL) return;
    int m = idx / DMODEL;
    int d = idx - m * DMODEL;
    x[idx] = embed[(size_t)ids[m] * DMODEL + d];
}

// =====================================================================
// fp32 -> bf16 bulk convert (weights, once per launch per matrix)
// =====================================================================
__global__ void f32_to_bf16_kernel(const float* __restrict__ in,
                                   bf16_t* __restrict__ out, int n) {
    int idx = (blockIdx.x * blockDim.x + threadIdx.x) * 4;
    if (idx + 3 >= n) {
        for (int i = idx; i < n; ++i) out[i] = (bf16_t)in[i];
        return;
    }
    float4 v = *(const float4*)(in + idx);
    out[idx + 0] = (bf16_t)v.x;
    out[idx + 1] = (bf16_t)v.y;
    out[idx + 2] = (bf16_t)v.z;
    out[idx + 3] = (bf16_t)v.w;
}

// =====================================================================
// RMSNorm -> bf16:  out[m,:] = x[m,:] * rsqrt(mean(x^2)+eps) * w
// =====================================================================
__global__ __launch_bounds__(256)
void rmsnorm_bf16_kernel(const float* __restrict__ x,
                         const float* __restrict__ w,
                         bf16_t* __restrict__ out) {
    __shared__ float red[256];
    int row = blockIdx.x;
    const float* xr = x + (size_t)row * DMODEL;
    float s = 0.f;
    for (int i = threadIdx.x; i < DMODEL; i += 256) {
        float v = xr[i];
        s += v * v;
    }
    red[threadIdx.x] = s;
    __syncthreads();
    for (int st = 128; st > 0; st >>= 1) {
        if (threadIdx.x < st) red[threadIdx.x] += red[threadIdx.x + st];
        __syncthreads();
    }
    float inv = rsqrtf(red[0] * (1.0f / DMODEL) + 1e-6f);
    for (int i = threadIdx.x; i < DMODEL; i += 256)
        out[(size_t)row * DMODEL + i] = (bf16_t)(xr[i] * inv * w[i]);
}

// =====================================================================
// WMMA GEMM (all-bf16 operands):
//   C[M,N] (+)= A_bf16[M,K] * W_bf16[N,K]^T + bias
// wave computes a 32x64 tile (2 M-subtiles x 4 N-subtiles = 8 WMMAs/step)
// A-frag (16x32): lane m = lane&15; contiguous K chunk of 16 at (lane>>4)*16
// B-frag (32x16): lane n = lane&15; contiguous K chunk of 16 at (lane>>4)*16
// C-frag (16x16): row = r + (lane>>4)*8, col = lane&15
// =====================================================================
__global__ __launch_bounds__(256)
void gemm_bf16_wmma(const bf16_t* __restrict__ A,
                    const bf16_t* __restrict__ W,
                    const float*  __restrict__ bias,
                    float* __restrict__ C,
                    int M, int N, int K, int accumulate, int tiles) {
    int wid = (blockIdx.x << 3) + (threadIdx.x >> 5);
    if (wid >= tiles) return;
    int tilesN = N >> 6;
    int tm = wid / tilesN;
    int tn = wid - tm * tilesN;

    int lane = threadIdx.x & 31;
    int hlf  = lane >> 4;
    int l16  = lane & 15;

    floatx8 acc[2][4] = {};

    const bf16_t* arow0 = A + (size_t)(tm * 32 + l16) * K + hlf * 16;
    const bf16_t* arow1 = arow0 + (size_t)16 * K;
    const bf16_t* wrow[4];
#pragma unroll
    for (int t = 0; t < 4; ++t)
        wrow[t] = W + (size_t)(tn * 64 + t * 16 + l16) * K + hlf * 16;

    for (int k = 0; k < K; k += 32) {
        // prefetch upcoming weight lines into cache (global_prefetch_b8)
        __builtin_prefetch(wrow[0] + k + 256, 0, 0);
        __builtin_prefetch(wrow[2] + k + 256, 0, 0);

        bf16x8 a0l = *(const bf16x8*)(arow0 + k);
        bf16x8 a0h = *(const bf16x8*)(arow0 + k + 8);
        bf16x8 a1l = *(const bf16x8*)(arow1 + k);
        bf16x8 a1h = *(const bf16x8*)(arow1 + k + 8);
        bf16x16 af0 = CAT16(a0l, a0h);
        bf16x16 af1 = CAT16(a1l, a1h);

#pragma unroll
        for (int t = 0; t < 4; ++t) {
            bf16x16 bf = *(const bf16x16*)(wrow[t] + k);
            acc[0][t] = __builtin_amdgcn_wmma_f32_16x16x32_bf16(
                false, af0, false, bf, (short)0, acc[0][t], false, false);
            acc[1][t] = __builtin_amdgcn_wmma_f32_16x16x32_bf16(
                false, af1, false, bf, (short)0, acc[1][t], false, false);
        }
    }

#pragma unroll
    for (int sm = 0; sm < 2; ++sm) {
#pragma unroll
        for (int t = 0; t < 4; ++t) {
            int col = tn * 64 + t * 16 + l16;
            float bv = bias ? bias[col] : 0.f;
#pragma unroll
            for (int r = 0; r < 8; ++r) {
                int row = tm * 32 + sm * 16 + r + hlf * 8;
                size_t ci = (size_t)row * N + col;
                float v = acc[sm][t][r] + bv;
                if (accumulate) v += C[ci];
                C[ci] = v;
            }
        }
    }
}

// =====================================================================
// RoPE for Q: q_f32 [M, D] -> q_bf16 [B, H, T, HS]
// =====================================================================
__global__ void rope_q_kernel(const float* __restrict__ q,
                              bf16_t* __restrict__ qbf) {
    int idx = blockIdx.x * blockDim.x + threadIdx.x;
    if (idx >= MTOK * NHEAD * 32) return;
    int i = idx & 31;
    int h = (idx >> 5) & (NHEAD - 1);
    int m = idx >> (5 + 4);
    int t = m & (SEQ - 1);
    int b = m >> 9;

    size_t src = (size_t)m * DMODEL + h * HEADDIM + i;
    float x1 = q[src];
    float x2 = q[src + 32];
    float ang = (float)t * powf(10000.f, -(float)(2 * i) / 64.f);
    float c = cosf(ang), sn = sinf(ang);
    size_t base = (((size_t)b * NHEAD + h) * SEQ + t) * HEADDIM;
    qbf[base + i]      = (bf16_t)(x1 * c - x2 * sn);
    qbf[base + i + 32] = (bf16_t)(x2 * c + x1 * sn);
}

// =====================================================================
// RoPE for K + pack V: kv_f32 [M, 2*G*HS] -> k_bf16/v_bf16 [B, G, T, HS]
// =====================================================================
__global__ void rope_kv_kernel(const float* __restrict__ kv,
                               bf16_t* __restrict__ kbf,
                               bf16_t* __restrict__ vbf) {
    int idx = blockIdx.x * blockDim.x + threadIdx.x;
    if (idx >= MTOK * NKV * 32) return;
    int i = idx & 31;
    int g = (idx >> 5) & (NKV - 1);
    int m = idx >> (5 + 2);
    int t = m & (SEQ - 1);
    int b = m >> 9;

    const float* kvm = kv + (size_t)m * KVDIM;
    float k1 = kvm[g * HEADDIM + i];
    float k2 = kvm[g * HEADDIM + i + 32];
    float v1 = kvm[NKV * HEADDIM + g * HEADDIM + i];
    float v2 = kvm[NKV * HEADDIM + g * HEADDIM + i + 32];

    float ang = (float)t * powf(10000.f, -(float)(2 * i) / 64.f);
    float c = cosf(ang), sn = sinf(ang);
    size_t base = (((size_t)b * NKV + g) * SEQ + t) * HEADDIM;
    kbf[base + i]      = (bf16_t)(k1 * c - k2 * sn);
    kbf[base + i + 32] = (bf16_t)(k2 * c + k1 * sn);
    vbf[base + i]      = (bf16_t)v1;
    vbf[base + i + 32] = (bf16_t)v2;
}

// =====================================================================
// Flash-style causal attention, all-WMMA, TDM-staged V tiles.
// block = 8 waves; each wave owns 16 query rows of one (b,h).
// Per 32-key tile: issue TENSOR_LOAD_TO_LDS for the 32x64 bf16 V tile
// (overlapped with the Q*K^T WMMAs), online softmax via lane shuffles,
// P restaged through LDS into an A-fragment, P*V from the LDS V tile.
// =====================================================================
__global__ __launch_bounds__(256)
void attn_fa_wmma(const bf16_t* __restrict__ Q,
                  const bf16_t* __restrict__ Kc,
                  const bf16_t* __restrict__ Vc,
                  bf16_t* __restrict__ O) {
    __shared__ __align__(16) bf16_t vtile[8][32 * HEADDIM];  // 32 KiB
    __shared__ __align__(16) bf16_t pbuf[8][16 * 32];        // 8 KiB

    const int qchunks = SEQ / 128;  // 4
    int bh = blockIdx.x / qchunks;
    int qc = blockIdx.x - bh * qchunks;
    int b = bh / NHEAD;
    int h = bh - b * NHEAD;
    int g = h / (NHEAD / NKV);

    int wid  = threadIdx.x >> 5;
    int lane = threadIdx.x & 31;
    int hlf  = lane >> 4;
    int l16  = lane & 15;
    int qbase = qc * 128 + wid * 16;

    const bf16_t* Qb = Q  + ((size_t)(b * NHEAD + h) * SEQ) * HEADDIM;
    const bf16_t* Kb = Kc + ((size_t)(b * NKV  + g) * SEQ) * HEADDIM;
    const bf16_t* Vb = Vc + ((size_t)(b * NKV  + g) * SEQ) * HEADDIM;

    // LDS byte offset of this wave's V staging buffer (LDS aperture
    // truncates generic addresses to addr[31:0]).
    unsigned vlds = (unsigned)(uintptr_t)(&vtile[wid][0]);

    float mrun[8], lrun[8];
#pragma unroll
    for (int r = 0; r < 8; ++r) { mrun[r] = -1e30f; lrun[r] = 0.f; }
    floatx8 oacc[4] = {};

    const float scale = 0.125f;  // 1/sqrt(64)
    int kend = qbase + 16;       // causal horizon for this wave

    for (int kt = 0; kt < kend; kt += 32) {
        // order: previous LDS reads must complete before TDM overwrites
        asm volatile("s_wait_dscnt 0" ::: "memory");

        // ---- issue TDM: V[kt..kt+32) x [0..64) bf16 -> LDS ----
        {
            unsigned long long ga =
                (unsigned long long)(uintptr_t)(Vb + (size_t)kt * HEADDIM);
            uintx4 g0;
            g0[0] = 1u;            // count=1, user mode
            g0[1] = vlds;          // lds_addr (bytes)
            g0[2] = (unsigned)ga;  // global_addr[31:0]
            g0[3] = (unsigned)((ga >> 32) & 0x01FFFFFFu) | (2u << 30);  // type=2
            intx8 g1;
            g1[0] = 0x00010000;                      // data_size=1 -> 2 bytes
            g1[1] = (int)((HEADDIM & 0xFFFF) << 16); // tensor_dim0 = 64 (lo16)
            g1[2] = (int)((SEQ & 0xFFFF) << 16);     // tdim0 hi=0 | tensor_dim1=512 lo
            g1[3] = (int)(64u << 16);                // tdim1 hi=0 | tile_dim0=64
            g1[4] = 32;                              // tile_dim1=32, tile_dim2=0
            g1[5] = HEADDIM;                         // tensor_dim0_stride = 64
            g1[6] = 0;
            g1[7] = 0;
            intx4 gz4 = {0, 0, 0, 0};
            intx8 gz8 = {0, 0, 0, 0, 0, 0, 0, 0};
            __builtin_amdgcn_tensor_load_to_lds(g0, g1, gz4, gz4, gz8, 0);
        }

        // ---- S = Q * K^T over this 32-key tile (overlaps the TDM) ----
        floatx8 s[2] = {};
#pragma unroll
        for (int kk = 0; kk < HEADDIM; kk += 32) {
            const bf16_t* qrow = Qb + (size_t)(qbase + l16) * HEADDIM + kk + hlf * 16;
            bf16x8 q0 = *(const bf16x8*)(qrow);
            bf16x8 q1 = *(const bf16x8*)(qrow + 8);
            bf16x16 af = CAT16(q0, q1);
#pragma unroll
            for (int j = 0; j < 2; ++j) {
                const bf16_t* krow =
                    Kb + (size_t)(kt + j * 16 + l16) * HEADDIM + kk + hlf * 16;
                bf16x16 bf = *(const bf16x16*)(krow);
                s[j] = __builtin_amdgcn_wmma_f32_16x16x32_bf16(
                    false, af, false, bf, (short)0, s[j], false, false);
            }
        }

        // ---- scale + causal mask + online softmax stats ----
        float mnew[8];
#pragma unroll
        for (int r = 0; r < 8; ++r) {
            int row = qbase + r + hlf * 8;
            float v0 = s[0][r] * scale;
            float v1 = s[1][r] * scale;
            if (kt + l16 > row)      v0 = -1e30f;
            if (kt + 16 + l16 > row) v1 = -1e30f;
            s[0][r] = v0; s[1][r] = v1;
            float mr = fmaxf(v0, v1);
            mr = fmaxf(mr, __shfl_xor(mr, 1));
            mr = fmaxf(mr, __shfl_xor(mr, 2));
            mr = fmaxf(mr, __shfl_xor(mr, 4));
            mr = fmaxf(mr, __shfl_xor(mr, 8));
            mnew[r] = fmaxf(mrun[r], mr);
        }
#pragma unroll
        for (int r = 0; r < 8; ++r) {
            float p0 = expf(s[0][r] - mnew[r]);
            float p1 = expf(s[1][r] - mnew[r]);
            float rs = p0 + p1;
            rs += __shfl_xor(rs, 1);
            rs += __shfl_xor(rs, 2);
            rs += __shfl_xor(rs, 4);
            rs += __shfl_xor(rs, 8);
            float alpha = expf(mrun[r] - mnew[r]);
            lrun[r] = lrun[r] * alpha + rs;
            mrun[r] = mnew[r];
#pragma unroll
            for (int t = 0; t < 4; ++t) oacc[t][r] *= alpha;
            int rowl = r + hlf * 8;
            pbuf[wid][rowl * 32 + l16]      = (bf16_t)p0;
            pbuf[wid][rowl * 32 + 16 + l16] = (bf16_t)p1;
        }

        // wait: TDM V tile landed; pbuf writes are in-order with ds reads
        __builtin_amdgcn_s_wait_tensorcnt(0);
        asm volatile("" ::: "memory");

        // ---- O += P * V from LDS ----
        bf16x8 p0v = *(const bf16x8*)(&pbuf[wid][l16 * 32 + hlf * 16]);
        bf16x8 p1v = *(const bf16x8*)(&pbuf[wid][l16 * 32 + hlf * 16 + 8]);
        bf16x16 pa = CAT16(p0v, p1v);
#pragma unroll
        for (int t = 0; t < 4; ++t) {
            bf16x16 vb;
#pragma unroll
            for (int i = 0; i < 16; ++i)
                vb[i] = vtile[wid][(hlf * 16 + i) * HEADDIM + t * 16 + l16];
            oacc[t] = __builtin_amdgcn_wmma_f32_16x16x32_bf16(
                false, pa, false, vb, (short)0, oacc[t], false, false);
        }
    }

    // ---- normalize and write out as [B, T, H*HS] bf16 ----
#pragma unroll
    for (int t = 0; t < 4; ++t) {
#pragma unroll
        for (int r = 0; r < 8; ++r) {
            int row = qbase + r + hlf * 8;
            float v = oacc[t][r] / lrun[r];
            O[((size_t)(b * SEQ + row)) * DMODEL + h * HEADDIM + t * 16 + l16] =
                (bf16_t)v;
        }
    }
}

// =====================================================================
// SwiGLU elementwise: out = bf16( silu(gate) * up )
// =====================================================================
__global__ void silu_mul_kernel(const float* __restrict__ gate,
                                const float* __restrict__ up,
                                bf16_t* __restrict__ out) {
    int idx = blockIdx.x * blockDim.x + threadIdx.x;
    if (idx >= MTOK * FFDIM) return;
    float gv = gate[idx];
    float sg = gv / (1.f + expf(-gv));
    out[idx] = (bf16_t)(sg * up[idx]);
}

// =====================================================================
// host-side launchers
// =====================================================================
static inline void launch_convert(const float* w, bf16_t* wbf, int n,
                                  hipStream_t s) {
    int threads = (n + 3) / 4;
    f32_to_bf16_kernel<<<(threads + 255) / 256, 256, 0, s>>>(w, wbf, n);
}

static inline void launch_gemm(const bf16_t* A, const bf16_t* W, const float* bias,
                               float* C, int M, int N, int K, int acc,
                               hipStream_t s) {
    int tiles = (M / 32) * (N / 64);
    int blocks = (tiles + 7) / 8;
    gemm_bf16_wmma<<<blocks, 256, 0, s>>>(A, W, bias, C, M, N, K, acc, tiles);
}

extern "C" void kernel_launch(void* const* d_in, const int* in_sizes, int n_in,
                              void* d_out, int out_size, void* d_ws, size_t ws_size,
                              hipStream_t stream) {
    (void)in_sizes; (void)n_in; (void)out_size; (void)ws_size;

    const int*   ids     = (const int*)  d_in[0];
    const float* embed   = (const float*)d_in[1];
    const float* q_w     = (const float*)d_in[2];
    const float* q_b     = (const float*)d_in[3];
    const float* kv_w    = (const float*)d_in[4];
    const float* kv_b    = (const float*)d_in[5];
    const float* o_w     = (const float*)d_in[6];
    const float* ln1_w   = (const float*)d_in[7];
    const float* ln2_w   = (const float*)d_in[8];
    const float* gate_w  = (const float*)d_in[9];
    const float* up_w    = (const float*)d_in[10];
    const float* down_w  = (const float*)d_in[11];
    const float* last_w  = (const float*)d_in[12];
    float* logits = (float*)d_out;

    char* ws = (char*)d_ws;
    size_t off = 0;
    float*  x      = (float*) (ws + off); off += (size_t)MTOK * DMODEL * 4;
    bf16_t* hbuf   = (bf16_t*)(ws + off); off += (size_t)MTOK * DMODEL * 2;
    float*  qf     = (float*) (ws + off); off += (size_t)MTOK * DMODEL * 4;
    float*  kvf    = (float*) (ws + off); off += (size_t)MTOK * KVDIM  * 4;
    bf16_t* qbf    = (bf16_t*)(ws + off); off += (size_t)MTOK * DMODEL * 2;
    bf16_t* kbf    = (bf16_t*)(ws + off); off += (size_t)BATCH * NKV * SEQ * HEADDIM * 2;
    bf16_t* vbf    = (bf16_t*)(ws + off); off += (size_t)BATCH * NKV * SEQ * HEADDIM * 2;
    bf16_t* attnbf = (bf16_t*)(ws + off); off += (size_t)MTOK * DMODEL * 2;
    float*  gatef  = (float*) (ws + off); off += (size_t)MTOK * FFDIM * 4;
    float*  upf    = (float*) (ws + off); off += (size_t)MTOK * FFDIM * 4;
    bf16_t* actbf  = (bf16_t*)(ws + off); off += (size_t)MTOK * FFDIM * 2;
    bf16_t* wbf    = (bf16_t*)(ws + off); off += (size_t)VOCAB * DMODEL * 2;  // weight staging

    // x = embed[ids]
    embed_gather_kernel<<<(MTOK * DMODEL + 255) / 256, 256, 0, stream>>>(ids, embed, x);

    for (int l = 0; l < LAYERS; ++l) {
        // h = rmsnorm(x, ln1)
        rmsnorm_bf16_kernel<<<MTOK, 256, 0, stream>>>(x, ln1_w + (size_t)l * DMODEL, hbuf);

        // q = h @ q_w^T + q_b
        launch_convert(q_w + (size_t)l * DMODEL * DMODEL, wbf, DMODEL * DMODEL, stream);
        launch_gemm(hbuf, wbf, q_b + (size_t)l * DMODEL, qf, MTOK, DMODEL, DMODEL, 0, stream);
        // kv = h @ kv_w^T + kv_b
        launch_convert(kv_w + (size_t)l * KVDIM * DMODEL, wbf, KVDIM * DMODEL, stream);
        launch_gemm(hbuf, wbf, kv_b + (size_t)l * KVDIM, kvf, MTOK, KVDIM, DMODEL, 0, stream);

        // RoPE + repack to [B,H,T,HS] / [B,G,T,HS] bf16
        rope_q_kernel<<<(MTOK * NHEAD * 32 + 255) / 256, 256, 0, stream>>>(qf, qbf);
        rope_kv_kernel<<<(MTOK * NKV * 32 + 255) / 256, 256, 0, stream>>>(kvf, kbf, vbf);

        // causal GQA attention (TDM-staged V)
        attn_fa_wmma<<<BATCH * NHEAD * (SEQ / 128), 256, 0, stream>>>(qbf, kbf, vbf, attnbf);

        // x += attn @ o_w^T
        launch_convert(o_w + (size_t)l * DMODEL * DMODEL, wbf, DMODEL * DMODEL, stream);
        launch_gemm(attnbf, wbf, nullptr, x, MTOK, DMODEL, DMODEL, 1, stream);

        // h = rmsnorm(x, ln2)
        rmsnorm_bf16_kernel<<<MTOK, 256, 0, stream>>>(x, ln2_w + (size_t)l * DMODEL, hbuf);

        // gate/up projections
        launch_convert(gate_w + (size_t)l * FFDIM * DMODEL, wbf, FFDIM * DMODEL, stream);
        launch_gemm(hbuf, wbf, nullptr, gatef, MTOK, FFDIM, DMODEL, 0, stream);
        launch_convert(up_w + (size_t)l * FFDIM * DMODEL, wbf, FFDIM * DMODEL, stream);
        launch_gemm(hbuf, wbf, nullptr, upf, MTOK, FFDIM, DMODEL, 0, stream);

        // silu(gate) * up
        silu_mul_kernel<<<(MTOK * FFDIM + 255) / 256, 256, 0, stream>>>(gatef, upf, actbf);

        // x += act @ down_w^T
        launch_convert(down_w + (size_t)l * DMODEL * FFDIM, wbf, DMODEL * FFDIM, stream);
        launch_gemm(actbf, wbf, nullptr, x, MTOK, DMODEL, FFDIM, 1, stream);
    }

    // final norm + tied lm_head (embed as bf16)
    rmsnorm_bf16_kernel<<<MTOK, 256, 0, stream>>>(x, last_w, hbuf);
    launch_convert(embed, wbf, VOCAB * DMODEL, stream);
    launch_gemm(hbuf, wbf, nullptr, logits, MTOK, VOCAB, DMODEL, 0, stream);
}
